// CausalAttentionHead_31714038514051
// MI455X (gfx1250) — matrix operand order repaired and checked by reading, not compiled, not run on hardware
//
#include <hip/hip_runtime.h>

// CausalAttentionHead for MI455X (gfx1250, wave32, WMMA bf16 16x16x32).
// Workspace layout (needs ~12.4 MiB):
//   [0,        4Mi)  qbf  [B,S,64] bf16, pre-scaled by 1/64
//   [4Mi,      8Mi)  kbf  [B,S,64] bf16
//   [8Mi,     12Mi)  vT   [B,64,S] bf16 (transposed for PV B-fragments)
//   [12Mi, 12.4Mi)  WT   [3,64,1024] bf16 (Wq/Wk/Wv transposed)

#define BB 16
#define SS 2048
#define DD 1024
#define HH 64

typedef __attribute__((ext_vector_type(16))) __bf16 v16bf;
typedef __attribute__((ext_vector_type(8)))  __bf16 v8bf;
typedef __attribute__((ext_vector_type(8)))  float  v8f;
typedef __attribute__((ext_vector_type(4)))  float  v4f;

static __device__ __forceinline__ v16bf cat16(v8bf lo, v8bf hi) {
  v16bf r;
#pragma unroll
  for (int i = 0; i < 8; ++i) { r[i] = lo[i]; r[i + 8] = hi[i]; }
  return r;
}

static __device__ __forceinline__ v8f wmma_bf16(v16bf a, v16bf b, v8f c) {
  // (neg_a, A, neg_b, B, c_mod, C, reuse_a, reuse_b)
  return __builtin_amdgcn_wmma_f32_16x16x32_bf16(false, a, false, b, (short)0, c,
                                                 false, false);
}

// ---------------------------------------------------------------------------
// Kernel 0: W [1024,64] f32 -> WT [3][64][1024] bf16 (transpose + convert)
// ---------------------------------------------------------------------------
__global__ void wt_convert_kernel(const float* __restrict__ Wq,
                                  const float* __restrict__ Wk,
                                  const float* __restrict__ Wv,
                                  __bf16* __restrict__ WT) {
  int idx = blockIdx.x * blockDim.x + threadIdx.x;  // exact grid: 3*64*1024
  int m3  = idx / (HH * DD);
  int rem = idx - m3 * (HH * DD);
  int h   = rem / DD;
  int d   = rem - h * DD;
  const float* W = (m3 == 0) ? Wq : ((m3 == 1) ? Wk : Wv);
  WT[idx] = (__bf16)W[(size_t)d * HH + h];
}

// ---------------------------------------------------------------------------
// Kernel 1: fused QKV projection. One wave per 16-row tile of x.
// x read exactly once (HBM-bound part). 12 WMMAs per K-step of 32.
// Both operand streams are software-pipelined one full K-step ahead:
//   - A fragment (x, HBM): next step's chunks load under this step's WMMAs.
//   - B fragments (WT, L2): rotate through bw[12]; each bw[i] is reloaded
//     with the next step's fragment right after its WMMA consumes it, giving
//     every load ~a full iteration of latency cover.
// ---------------------------------------------------------------------------
__global__ __launch_bounds__(32) void qkv_proj_kernel(
    const float* __restrict__ x, const __bf16* __restrict__ WT,
    __bf16* __restrict__ qbf, __bf16* __restrict__ kbf,
    __bf16* __restrict__ vT) {
  const int lane = threadIdx.x;
  const int m16  = lane & 15;   // row (A) / column (B,C,D) within tile
  const int hl   = lane >> 4;   // half-wave group
  const int t0   = blockIdx.x;
  const int b    = t0 / (SS / 16);
  const int s0   = (t0 % (SS / 16)) * 16;

  const float* xrow = x + ((size_t)b * SS + s0 + m16) * DD;
  v8f acc[3][4] = {};

  // prime the A-fragment pipeline
  v4f xa0 = *(const v4f*)(xrow + hl * 16);
  v4f xa1 = *(const v4f*)(xrow + hl * 16 + 4);
  v4f xa2 = *(const v4f*)(xrow + hl * 16 + 8);
  v4f xa3 = *(const v4f*)(xrow + hl * 16 + 12);

  // prime the B-fragment pipeline: 12 fragments (3 matrices x 4 N-tiles).
  // B fragment (32x16): lane = col, rows kk+hl*16+[0..15], contiguous 32B.
  const __bf16* wbase = WT + (size_t)m16 * DD + hl * 16;
  v16bf bw[12];
#pragma unroll
  for (int i = 0; i < 12; ++i)
    bw[i] = *(const v16bf*)(wbase + (size_t)((i / 4) * 64 + (i % 4) * 16) * DD);

  for (int kk = 0; kk < DD; kk += 32) {
    // A fragment (16x32 bf16): lane holds K = hl*16+[0..15]
    v16bf a;
#pragma unroll
    for (int e = 0; e < 4; ++e) {
      a[e]      = (__bf16)xa0[e];
      a[4 + e]  = (__bf16)xa1[e];
      a[8 + e]  = (__bf16)xa2[e];
      a[12 + e] = (__bf16)xa3[e];
    }
    // issue next step's x loads now; they retire under the WMMA block below
    if (kk + 32 < DD) {
      const float* nx = xrow + kk + 32;
      xa0 = *(const v4f*)(nx + hl * 16);
      xa1 = *(const v4f*)(nx + hl * 16 + 4);
      xa2 = *(const v4f*)(nx + hl * 16 + 8);
      xa3 = *(const v4f*)(nx + hl * 16 + 12);
      __builtin_prefetch(xrow + kk + 32 + 256, 0, 1);  // stream x ~8 steps ahead
    }
    const int kn = (kk + 32 < DD) ? kk + 32 : 0;  // clamped: stays in-bounds
#pragma unroll
    for (int i = 0; i < 12; ++i) {
      acc[i / 4][i % 4] = wmma_bf16(a, bw[i], acc[i / 4][i % 4]);
      // bw[i] is dead: refill with next K-step's fragment (WAR keeps order)
      bw[i] = *(const v16bf*)(wbase +
                              (size_t)((i / 4) * 64 + (i % 4) * 16) * DD + kn);
    }
  }

  // C/D layout: vgpr r, lane -> row m = r + 8*hl, col n = m16
#pragma unroll
  for (int nt = 0; nt < 4; ++nt) {
#pragma unroll
    for (int r = 0; r < 8; ++r) {
      const int m = r + 8 * hl;
      const int h = nt * 16 + m16;
      const size_t row = (size_t)b * SS + s0 + m;
      qbf[row * HH + h] = (__bf16)(acc[0][nt][r] * (1.0f / 64.0f));  // fold /H
      kbf[row * HH + h] = (__bf16)(acc[1][nt][r]);
      vT[((size_t)b * HH + h) * SS + s0 + m] = (__bf16)(acc[2][nt][r]);
    }
  }
}

// ---------------------------------------------------------------------------
// Kernel 2: flash attention, one wave per 16-row q tile. K/V stream from L2.
// Pairs of 16-key tiles -> K=32 for the PV WMMA. Causal mask also zeroes the
// phantom tile when the tile count is odd (no divergent branches -> EXEC stays
// all-ones around WMMA). Pipelining: next pair's K fragments load right after
// the QK^T WMMAs consume the current ones, and this pair's V fragments load
// before the softmax, so the VALU-heavy softmax hides both L2 streams.
// ---------------------------------------------------------------------------
__global__ __launch_bounds__(32) void attn_fa_kernel(
    const __bf16* __restrict__ qbf, const __bf16* __restrict__ kbf,
    const __bf16* __restrict__ vT, float* __restrict__ out) {
  __shared__ __align__(32) __bf16 pL[16 * 32];  // P tile staging (D -> A layout)

  const int lane = threadIdx.x;
  const int m16  = lane & 15;
  const int hl   = lane >> 4;
  const int t0   = blockIdx.x;
  const int b    = t0 / (SS / 16);
  const int s0   = (t0 % (SS / 16)) * 16;

  // Q as two 16x32 A-fragments (head dims 0..31 and 32..63), loaded once.
  const __bf16* qrow = qbf + ((size_t)b * SS + s0 + m16) * HH;
  v16bf aq[2];
#pragma unroll
  for (int f = 0; f < 2; ++f) {
    v8bf c0 = *(const v8bf*)(qrow + f * 32 + hl * 16);
    v8bf c1 = *(const v8bf*)(qrow + f * 32 + hl * 16 + 8);
    aq[f] = cat16(c0, c1);
  }

  float mrow[8], lrow[8];
  v8f o[4] = {};
#pragma unroll
  for (int r = 0; r < 8; ++r) { mrow[r] = -__builtin_inff(); lrow[r] = 0.0f; }

  v16bf bk[2][2];  // [subtile u][head-dim fragment]
  auto load_k = [&](int jtt) {
#pragma unroll
    for (int u = 0; u < 2; ++u) {
      const int key0 = (jtt + u) * 16;  // phantom tile stays in-bounds
      const __bf16* krow = kbf + ((size_t)b * SS + key0 + m16) * HH;
      bk[u][0] = *(const v16bf*)(krow + hl * 16);       // head dims 0..31
      bk[u][1] = *(const v16bf*)(krow + 32 + hl * 16);  // head dims 32..63
    }
  };

  const int ntile = s0 / 16 + 1;  // 16-key tiles needed under causal mask
  load_k(0);
  for (int jt = 0; jt < ntile; jt += 2) {
    // ---- scores for both 16-key subtiles (4 WMMAs) ----
    v8f sc[2];
#pragma unroll
    for (int u = 0; u < 2; ++u) {
      v8f s = {};
      s = wmma_bf16(aq[0], bk[u][0], s);
      s = wmma_bf16(aq[1], bk[u][1], s);
      sc[u] = s;
    }
    // bk is dead now: issue next pair's K loads; softmax below hides them.
    load_k((jt + 2 < ntile) ? jt + 2 : 0);
    // This pair's V fragments: issue now, consume after the LDS round-trip.
    const int key0p = jt * 16;
    v16bf bv[4];
#pragma unroll
    for (int nt = 0; nt < 4; ++nt)
      bv[nt] = *(const v16bf*)(
          vT + ((size_t)b * HH + nt * 16 + m16) * SS + key0p + hl * 16);
    __builtin_prefetch(vT + ((size_t)b * HH + m16) * SS + (jt + 2) * 16, 0, 1);

    // ---- causal mask + online softmax over 32 keys at once ----
#pragma unroll
    for (int r = 0; r < 8; ++r) {
      const int qi  = s0 + r + 8 * hl;
      const int ki0 = jt * 16 + m16;
      const int ki1 = ki0 + 16;
      float sv0 = (ki0 <= qi) ? sc[0][r] : -__builtin_inff();
      float sv1 = (ki1 <= qi) ? sc[1][r] : -__builtin_inff();
      float rm = fmaxf(sv0, sv1);
#pragma unroll
      for (int d = 1; d < 16; d <<= 1) rm = fmaxf(rm, __shfl_xor(rm, d, 32));
      const float mnew  = fmaxf(mrow[r], rm);
      const float alpha = __expf(mrow[r] - mnew);
      const float p0 = __expf(sv0 - mnew);
      const float p1 = __expf(sv1 - mnew);
      float rs = p0 + p1;
#pragma unroll
      for (int d = 1; d < 16; d <<= 1) rs += __shfl_xor(rs, d, 32);
      lrow[r] = lrow[r] * alpha + rs;
      mrow[r] = mnew;
#pragma unroll
      for (int nt = 0; nt < 4; ++nt) o[nt][r] *= alpha;
      const int m = r + 8 * hl;
      pL[m * 32 + m16]      = (__bf16)p0;
      pL[m * 32 + 16 + m16] = (__bf16)p1;
    }
    __syncthreads();

    // ---- PV: P (16x32 A-frag from LDS) x V (32x64, 4 B-frags) ----
    v8bf pc0 = *(const v8bf*)(pL + m16 * 32 + hl * 16);
    v8bf pc1 = *(const v8bf*)(pL + m16 * 32 + hl * 16 + 8);
    v16bf ap = cat16(pc0, pc1);
#pragma unroll
    for (int nt = 0; nt < 4; ++nt) o[nt] = wmma_bf16(ap, bv[nt], o[nt]);
    __syncthreads();
  }

  // ---- epilogue: normalize and store fp32 output ----
#pragma unroll
  for (int nt = 0; nt < 4; ++nt) {
#pragma unroll
    for (int r = 0; r < 8; ++r) {
      const int m = r + 8 * hl;
      out[((size_t)b * SS + s0 + m) * HH + nt * 16 + m16] = o[nt][r] / lrow[r];
    }
  }
}

// ---------------------------------------------------------------------------
extern "C" void kernel_launch(void* const* d_in, const int* in_sizes, int n_in,
                              void* d_out, int out_size, void* d_ws,
                              size_t ws_size, hipStream_t stream) {
  const float* x  = (const float*)d_in[0];
  const float* Wq = (const float*)d_in[1];
  const float* Wk = (const float*)d_in[2];
  const float* Wv = (const float*)d_in[3];
  float* out = (float*)d_out;

  char* ws = (char*)d_ws;
  const size_t nElem = (size_t)BB * SS * HH;  // 2 Mi elements per matrix
  __bf16* qbf = (__bf16*)(ws);
  __bf16* kbf = (__bf16*)(ws + 2 * nElem);
  __bf16* vT  = (__bf16*)(ws + 4 * nElem);
  __bf16* WT  = (__bf16*)(ws + 6 * nElem);  // 3*64*1024 bf16 = 384 KiB

  wt_convert_kernel<<<(3 * DD * HH) / 256, 256, 0, stream>>>(Wq, Wk, Wv, WT);
  qkv_proj_kernel<<<BB * (SS / 16), 32, 0, stream>>>(x, WT, qbf, kbf, vT);
  attn_fa_kernel<<<BB * (SS / 16), 32, 0, stream>>>(qbf, kbf, vT, out);

  (void)in_sizes; (void)n_in; (void)out_size; (void)ws_size;
}
